// SelfAttention_54082228191794
// MI455X (gfx1250) — matrix operand order repaired and checked by reading, not compiled, not run on hardware
//
#include <hip/hip_runtime.h>
#include <hip/hip_bf16.h>

// ---------------------------------------------------------------------------
// SelfAttention for MI455X (gfx1250, wave32, WMMA)
//   x:(2,256,128,128) f32 -> out = gamma * attn_out + x
//   Pool-first K/V (linear commutes with mean pool), bf16 WMMA w/ f32 accum,
//   flash-attention streaming softmax, double-buffered async-to-LDS V staging.
// ---------------------------------------------------------------------------

typedef __attribute__((ext_vector_type(16))) __bf16 v16bf;
typedef __attribute__((ext_vector_type(8)))  float  v8f;

union V16BF { uint4 u[2]; v16bf v; };

typedef __attribute__((address_space(3))) __bf16 lds_bf16_t;

#define BATCH   2
#define CDIM    256
#define C8DIM   32
#define NTOK    16384   // 128*128 query tokens
#define MTOK    4096    // 64*64 pooled key/value tokens

// ---------------------------------------------------------------------------
// Weight f32 -> bf16 conversion (Wq, Wk: 32x256; Wv: 256x256)
// ---------------------------------------------------------------------------
__global__ void wconv_kernel(const float* __restrict__ Wq,
                             const float* __restrict__ Wk,
                             const float* __restrict__ Wv,
                             __bf16* __restrict__ Wqb,
                             __bf16* __restrict__ Wkb,
                             __bf16* __restrict__ Wvb) {
    int i = blockIdx.x * blockDim.x + threadIdx.x;
    if (i < C8DIM * CDIM) { Wqb[i] = (__bf16)Wq[i]; Wkb[i] = (__bf16)Wk[i]; }
    if (i < CDIM * CDIM)  { Wvb[i] = (__bf16)Wv[i]; }
}

// ---------------------------------------------------------------------------
// LDS-tiled transpose+convert: x(b,c,t) f32 -> Xt(b,t,c) bf16  (token-major)
// block (32,8): 32c x 32t tile
// ---------------------------------------------------------------------------
__global__ void transpose_kernel(const float* __restrict__ x,
                                 __bf16* __restrict__ Xt) {
    __shared__ __bf16 tile[32][33];
    const int b  = blockIdx.z;
    const int t0 = blockIdx.x * 32;
    const int c0 = blockIdx.y * 32;
    const float* xb = x + ((size_t)b * CDIM + c0) * NTOK + t0;
#pragma unroll
    for (int i = 0; i < 4; ++i) {
        int c = threadIdx.y + 8 * i;
        tile[c][threadIdx.x] = (__bf16)xb[(size_t)c * NTOK + threadIdx.x];
    }
    __syncthreads();
    __bf16* yb = Xt + ((size_t)b * NTOK + t0) * CDIM + c0;
#pragma unroll
    for (int i = 0; i < 4; ++i) {
        int t = threadIdx.y + 8 * i;
        yb[(size_t)t * CDIM + threadIdx.x] = tile[threadIdx.x][t];
    }
}

// ---------------------------------------------------------------------------
// 2x2 mean pool + transpose: x(b,c,128,128) -> Xpt(b, s=4096, c) bf16
// ---------------------------------------------------------------------------
__global__ void pool_kernel(const float* __restrict__ x,
                            __bf16* __restrict__ Xpt) {
    int idx = blockIdx.x * blockDim.x + threadIdx.x;   // b*c*4096, sj fastest
    if (idx >= BATCH * CDIM * MTOK) return;
    int sj = idx & 63;
    int si = (idx >> 6) & 63;
    int c  = (idx >> 12) & 255;
    int b  = idx >> 20;
    const float* p = x + (((size_t)b * CDIM + c) * 128 + 2 * si) * 128 + 2 * sj;
    float m = 0.25f * (p[0] + p[1] + p[128] + p[129]);
    Xpt[((size_t)b * MTOK + (si * 64 + sj)) * CDIM + c] = (__bf16)m;
}

// ---------------------------------------------------------------------------
// Projection GEMM:  Y(tok, Cout) = Xin(tok,256) * W^T(256,Cout) + bias, *scale
// One wave per block; wave tile = 64 tokens x 16 couts (4 WMMA accum tiles).
// ---------------------------------------------------------------------------
__global__ __launch_bounds__(32)
void proj_kernel(const __bf16* __restrict__ Xin, const __bf16* __restrict__ Wb,
                 const float* __restrict__ bias, __bf16* __restrict__ Y,
                 int Mtok, int Cout, float scale, int cmajor) {
    const int b    = blockIdx.z;
    const int t0   = blockIdx.x * 64;
    const int n0   = blockIdx.y * 16;
    const int lane = threadIdx.x;
    const int half = lane >> 4;
    const int ln   = lane & 15;

    const __bf16* Xb = Xin + (size_t)b * Mtok * CDIM;
    __bf16*       Yb = Y   + (size_t)b * Mtok * Cout;

    const v8f vzero = {0.f, 0.f, 0.f, 0.f, 0.f, 0.f, 0.f, 0.f};
    v8f acc[4];
#pragma unroll
    for (int s = 0; s < 4; ++s) acc[s] = vzero;

#pragma unroll 2
    for (int k0 = 0; k0 < CDIM; k0 += 32) {
        // B operand (32x16): lane -> column n0+ln, 16 consecutive k
        V16BF bb;
        const __bf16* wp = Wb + (size_t)(n0 + ln) * CDIM + k0 + half * 16;
        bb.u[0] = ((const uint4*)wp)[0];
        bb.u[1] = ((const uint4*)wp)[1];
#pragma unroll
        for (int sub = 0; sub < 4; ++sub) {
            // A operand (16x32): lane -> row token, 16 consecutive k (h*16..+15)
            V16BF aa;
            const __bf16* xp = Xb + (size_t)(t0 + 16 * sub + ln) * CDIM + k0 + half * 16;
            aa.u[0] = ((const uint4*)xp)[0];
            aa.u[1] = *(const uint4*)(xp + 8);
            acc[sub] = __builtin_amdgcn_wmma_f32_16x16x32_bf16(
                false, aa.v, false, bb.v, (short)0, acc[sub], false, false);
        }
    }

    const float bval = bias[n0 + ln];
#pragma unroll
    for (int sub = 0; sub < 4; ++sub) {
#pragma unroll
        for (int j = 0; j < 8; ++j) {
            float v  = (acc[sub][j] + bval) * scale;
            int  tok = t0 + 16 * sub + j + 8 * half;
            int  ch  = n0 + ln;
            if (cmajor) Yb[(size_t)ch * Mtok + tok] = (__bf16)v;
            else        Yb[(size_t)tok * Cout + ch] = (__bf16)v;
        }
    }
}

// ---------------------------------------------------------------------------
// Flash attention: block = 128 threads (4 waves).  Each wave owns 16 query
// tokens x all 256 channels (16 f32 accumulator tiles).  Keys streamed in
// chunks of 32 with online softmax.  V chunks double-buffered in LDS via
// GLOBAL_LOAD_ASYNC_TO_LDS_B128 (ASYNCcnt), overlapped with WMMA compute.
//   Qt: (b, 16384, 32) bf16 (1/sqrt(32) pre-folded)
//   Kt: (b, 4096, 32)  bf16 token-major (score B columns contiguous in c8)
//   Vc: (b, 256, 4096) bf16 channel-major (PV B columns contiguous in keys)
// ---------------------------------------------------------------------------
__global__ __launch_bounds__(128)
void fa_kernel(const __bf16* __restrict__ Qt, const __bf16* __restrict__ Kt,
               const __bf16* __restrict__ Vc, const float* __restrict__ x,
               const float* __restrict__ gamma, float* __restrict__ out) {
    const int b    = blockIdx.y;
    const int lane = threadIdx.x & 31;
    const int wave = threadIdx.x >> 5;
    const int t0   = (blockIdx.x * 4 + wave) * 16;
    const int half = lane >> 4;
    const int ln   = lane & 15;

    const __bf16* Qb = Qt + (size_t)b * NTOK * C8DIM;
    const __bf16* Kb = Kt + (size_t)b * MTOK * C8DIM;
    const __bf16* Vb = Vc + (size_t)b * CDIM * MTOK;

    __shared__ __align__(16) __bf16 vbuf[2][CDIM][32];  // double-buffered V chunk
    __shared__ __align__(16) __bf16 pbuf[4][16][32];    // per-wave P transpose buffer

    // Issue async V-chunk load (256 rows x 64B); 2 rows per thread, 4x b128 each.
    // INST_OFFSET is added to both LDS and global address (ISA 10. async ops).
    auto issue_v_async = [&](int s0c, int bufi) {
        for (int ch = (int)threadIdx.x; ch < CDIM; ch += 128) {
            unsigned ldsoff = (unsigned)(size_t)(lds_bf16_t*)&vbuf[bufi][ch][0];
            unsigned long long ga =
                (unsigned long long)(Vb + (size_t)ch * MTOK + s0c);
            asm volatile(
                "global_load_async_to_lds_b128 %0, %1, off\n\t"
                "global_load_async_to_lds_b128 %0, %1, off offset:16\n\t"
                "global_load_async_to_lds_b128 %0, %1, off offset:32\n\t"
                "global_load_async_to_lds_b128 %0, %1, off offset:48"
                :: "v"(ldsoff), "v"(ga) : "memory");
        }
    };

    // Q A-operand: loaded once (16 tokens x 32 c8 = entire contraction depth)
    V16BF qa;
    {
        const __bf16* qrow = Qb + (size_t)(t0 + ln) * C8DIM + half * 16;
        qa.u[0] = ((const uint4*)qrow)[0];
        qa.u[1] = *(const uint4*)(qrow + 8);
    }

    const v8f vzero = {0.f, 0.f, 0.f, 0.f, 0.f, 0.f, 0.f, 0.f};
    v8f acc[16];
#pragma unroll
    for (int n = 0; n < 16; ++n) acc[n] = vzero;
    float mrun[8], lrun[8];
#pragma unroll
    for (int j = 0; j < 8; ++j) { mrun[j] = -INFINITY; lrun[j] = 0.f; }

    issue_v_async(0, 0);   // prologue: first chunk into buffer 0

    for (int s0 = 0, it = 0; s0 < MTOK; s0 += 32, ++it) {
        const int buf = it & 1;
        // Wait for this wave's async loads of the current buffer, then make
        // all waves' LDS writes visible block-wide.
        asm volatile("s_wait_asynccnt 0" ::: "memory");
        __syncthreads();
        // All waves finished reading buf^1 before this barrier -> safe to refill.
        if (s0 + 32 < MTOK) issue_v_async(s0 + 32, buf ^ 1);

        // Prefetch next K chunk (global_prefetch_b8)
        if (s0 + 32 < MTOK)
            __builtin_prefetch(Kb + (size_t)(s0 + 32 + lane) * C8DIM, 0, 2);

        // Score tiles: S = Q(16x32) x K(32x16) for keys [s0..s0+15], [s0+16..s0+31]
        V16BF kb0, kb1;
        {
            const __bf16* kc0 = Kb + (size_t)(s0 + ln) * C8DIM + half * 16;
            kb0.u[0] = ((const uint4*)kc0)[0];
            kb0.u[1] = ((const uint4*)kc0)[1];
            const __bf16* kc1 = kc0 + 16 * C8DIM;
            kb1.u[0] = ((const uint4*)kc1)[0];
            kb1.u[1] = ((const uint4*)kc1)[1];
        }
        v8f sv0 = __builtin_amdgcn_wmma_f32_16x16x32_bf16(
            false, qa.v, false, kb0.v, (short)0, vzero, false, false);
        v8f sv1 = __builtin_amdgcn_wmma_f32_16x16x32_bf16(
            false, qa.v, false, kb1.v, (short)0, vzero, false, false);

        // Online softmax per row (row m = j + 8*half lives in 16 lanes of a half)
        float corrv[8];
#pragma unroll
        for (int j = 0; j < 8; ++j) {
            float a0 = sv0[j], a1 = sv1[j];
            float rm = fmaxf(a0, a1);
            rm = fmaxf(rm, __shfl_xor(rm, 1));
            rm = fmaxf(rm, __shfl_xor(rm, 2));
            rm = fmaxf(rm, __shfl_xor(rm, 4));
            rm = fmaxf(rm, __shfl_xor(rm, 8));
            float mnew = fmaxf(mrun[j], rm);
            float corr = __expf(mrun[j] - mnew);
            float p0 = __expf(a0 - mnew);
            float p1 = __expf(a1 - mnew);
            float rs = p0 + p1;
            rs += __shfl_xor(rs, 1);
            rs += __shfl_xor(rs, 2);
            rs += __shfl_xor(rs, 4);
            rs += __shfl_xor(rs, 8);
            lrun[j] = lrun[j] * corr + rs;
            mrun[j] = mnew;
            corrv[j] = corr;
            // C-layout -> row-major P in LDS (per-wave region)
            pbuf[wave][j + 8 * half][ln]      = (__bf16)p0;
            pbuf[wave][j + 8 * half][ln + 16] = (__bf16)p1;
        }
        // Rescale accumulators by exp(m_old - m_new)
#pragma unroll
        for (int j = 0; j < 8; ++j) {
            float c = corrv[j];
#pragma unroll
            for (int n = 0; n < 16; ++n) acc[n][j] *= c;
        }
        asm volatile("" ::: "memory");   // keep ds_load after ds_store (same-wave, in-order LDS)

        // P A-operand (16 tokens x 32 keys) from LDS in WMMA A layout
        V16BF pa;
        pa.u[0] = *(const uint4*)(&pbuf[wave][ln][half * 16]);
        pa.u[1] = *(const uint4*)(&pbuf[wave][ln][half * 16 + 8]);

        // acc[n] += P(16x32) x Vchunk(32 x 16ch)   for 16 channel blocks
#pragma unroll
        for (int n = 0; n < 16; ++n) {
            V16BF vb;
            const __bf16* vrow = &vbuf[buf][16 * n + ln][half * 16];
            vb.u[0] = ((const uint4*)vrow)[0];
            vb.u[1] = ((const uint4*)vrow)[1];
            acc[n] = __builtin_amdgcn_wmma_f32_16x16x32_bf16(
                false, pa.v, false, vb.v, (short)0, acc[n], false, false);
        }
    }

    // Epilogue: out[b][c][t] = gamma * acc/l + x[b][c][t]
    const float g = gamma[0];
    const float* xb = x   + (size_t)b * CDIM * NTOK;
    float*       ob = out + (size_t)b * CDIM * NTOK;
#pragma unroll
    for (int j = 0; j < 8; ++j) {
        float inv = 1.0f / lrun[j];
        int tok = t0 + j + 8 * half;
#pragma unroll
        for (int n = 0; n < 16; ++n) {
            int ch = 16 * n + ln;
            size_t idx = (size_t)ch * NTOK + tok;
            ob[idx] = g * (acc[n][j] * inv) + xb[idx];
        }
    }
}

// ---------------------------------------------------------------------------
// Launch
// ---------------------------------------------------------------------------
extern "C" void kernel_launch(void* const* d_in, const int* in_sizes, int n_in,
                              void* d_out, int out_size, void* d_ws, size_t ws_size,
                              hipStream_t stream) {
    const float* x     = (const float*)d_in[0];
    const float* Wq    = (const float*)d_in[1];
    const float* bq    = (const float*)d_in[2];
    const float* Wk    = (const float*)d_in[3];
    const float* bk    = (const float*)d_in[4];
    const float* Wv    = (const float*)d_in[5];
    const float* bv    = (const float*)d_in[6];
    const float* gamma = (const float*)d_in[7];
    float* out = (float*)d_out;

    char* p = (char*)d_ws;
    auto carve = [&](size_t bytes) {
        char* r = p;
        p += (bytes + 255) & ~(size_t)255;
        return r;
    };
    __bf16* Xt  = (__bf16*)carve((size_t)BATCH * NTOK * CDIM * 2);   // 16 MB
    __bf16* Xpt = (__bf16*)carve((size_t)BATCH * MTOK * CDIM * 2);   //  4 MB
    __bf16* Qt  = (__bf16*)carve((size_t)BATCH * NTOK * C8DIM * 2);  //  2 MB
    __bf16* Kt  = (__bf16*)carve((size_t)BATCH * MTOK * C8DIM * 2);  // .5 MB
    __bf16* Vc  = (__bf16*)carve((size_t)BATCH * CDIM * MTOK * 2);   //  4 MB
    __bf16* Wqb = (__bf16*)carve((size_t)C8DIM * CDIM * 2);
    __bf16* Wkb = (__bf16*)carve((size_t)C8DIM * CDIM * 2);
    __bf16* Wvb = (__bf16*)carve((size_t)CDIM * CDIM * 2);

    const float qscale = 0.17677669529663689f;   // 1/sqrt(C8)

    wconv_kernel<<<(CDIM * CDIM + 255) / 256, 256, 0, stream>>>(Wq, Wk, Wv, Wqb, Wkb, Wvb);
    transpose_kernel<<<dim3(NTOK / 32, CDIM / 32, BATCH), dim3(32, 8), 0, stream>>>(x, Xt);
    pool_kernel<<<(BATCH * CDIM * MTOK) / 256, 256, 0, stream>>>(x, Xpt);

    // Q = (Wq Xt + bq)/sqrt(32), token-major
    proj_kernel<<<dim3(NTOK / 64, C8DIM / 16, BATCH), 32, 0, stream>>>(
        Xt, Wqb, bq, Qt, NTOK, C8DIM, qscale, 0);
    // K = Wk Xpt + bk, token-major
    proj_kernel<<<dim3(MTOK / 64, C8DIM / 16, BATCH), 32, 0, stream>>>(
        Xpt, Wkb, bk, Kt, MTOK, C8DIM, 1.0f, 0);
    // V = Wv Xpt + bv, channel-major
    proj_kernel<<<dim3(MTOK / 64, CDIM / 16, BATCH), 32, 0, stream>>>(
        Xpt, Wvb, bv, Vc, MTOK, CDIM, 1.0f, 1);

    fa_kernel<<<dim3(NTOK / 64, BATCH), 128, 0, stream>>>(Qt, Kt, Vc, x, gamma, out);
}
